// Decoder_18124761989253
// MI455X (gfx1250) — matrix operand order repaired and checked by reading, not compiled, not run on hardware
//
#include <hip/hip_runtime.h>

// Problem constants
#define B_    32
#define T_    64
#define S_    512
#define H_    512
#define E_    300
#define V_    32000
#define EPAD_ 320
#define K0C_  832     // EPAD_ + H_
#define K1C_  1024    // H_ + H_
#define G_    2048    // 4*H_
#define NEGV  (-1.0e12f)
#define NBLK_ 64      // persistent-kernel blocks (must be >= 36, <= resident capacity)

typedef __attribute__((ext_vector_type(16))) _Float16 v16h;
typedef __attribute__((ext_vector_type(8)))  float    v8f;
typedef __attribute__((ext_vector_type(4)))  float    v4f;

union AV { v16h v; v4f q[2]; };
union CV { v8f  v; float f[8]; };

__device__ __forceinline__ float sigmf(float x) { return 1.0f / (1.0f + __expf(-x)); }

// ---- WMMA tile loaders (CDNA5 16x16x32 f16 layouts, ISA 7.12.2) ----
// A: row-major [M,K]; lane m = L&15, k-half = 8*(L>>4); elems 0..7 -> k=kh..kh+7, 8..15 -> k=16+kh..
__device__ __forceinline__ v16h load_a16(const _Float16* __restrict__ A, int lda, int m0, int k0) {
  const int lane = threadIdx.x & 31;
  const int m  = m0 + (lane & 15);
  const int kh = (lane >> 4) << 3;            // 0 or 8
  const _Float16* p = A + (size_t)m * lda + k0 + kh;
  AV a;
  a.q[0] = *(const v4f*)(p);
  a.q[1] = *(const v4f*)(p + 16);
  return a.v;
}
// B: weights row-major [N,K] (i.e. B[k][n] = W[n][k]); lane n = L&15, k-half = 8*(L>>4); elems 0..7 -> k=kh..kh+7, 8..15 -> k=16+kh..
__device__ __forceinline__ v16h load_b16(const _Float16* __restrict__ W, int ldb, int n0, int k0) {
  const int lane = threadIdx.x & 31;
  const int n  = n0 + (lane & 15);
  const int kh = (lane >> 4) << 3;            // 0 or 8
  const _Float16* p = W + (size_t)n * ldb + k0 + kh;
  AV b;
  b.q[0] = *(const v4f*)(p);
  b.q[1] = *(const v4f*)(p + 16);
  return b.v;
}

__device__ __forceinline__ v8f wmma16(v16h a, v16h b, v8f c) {
  // (neg_a, A, neg_b, B, c_mod, C, reuse_a, reuse_b)
  return __builtin_amdgcn_wmma_f32_16x16x32_f16(false, a, false, b, (short)0, c, false, false);
}

// One 16x16 output tile, single wave, runtime K (multiple of 32).
__device__ __forceinline__ void gemm_tile(const _Float16* __restrict__ A, int lda,
                                          const _Float16* __restrict__ W, int ldb,
                                          float* __restrict__ C, int ldc,
                                          int m0, int n0, int K) {
  v8f acc = {};
  for (int k = 0; k < K; k += 32) {
    v16h a = load_a16(A, lda, m0, k);
    v16h b = load_b16(W, ldb, n0, k);
    acc = wmma16(a, b, acc);
  }
  const int lane = threadIdx.x & 31;
  const int n  = n0 + (lane & 15);
  const int mb = m0 + ((lane >> 4) << 3);
  CV c; c.v = acc;
#pragma unroll
  for (int r = 0; r < 8; ++r)
    C[(size_t)(mb + r) * ldc + n] = c.f[r];
}

// ---- device-scope sense barrier (state re-zeroed by prep each call) ----
__device__ __forceinline__ void gbarrier(unsigned* bar, unsigned nblk) {
  __syncthreads();
  __threadfence();
  if (threadIdx.x == 0) {
    unsigned gen = __hip_atomic_load(&bar[1], __ATOMIC_RELAXED, __HIP_MEMORY_SCOPE_AGENT);
    if (__hip_atomic_fetch_add(&bar[0], 1u, __ATOMIC_ACQ_REL, __HIP_MEMORY_SCOPE_AGENT) == nblk - 1u) {
      __hip_atomic_store(&bar[0], 0u, __ATOMIC_RELAXED, __HIP_MEMORY_SCOPE_AGENT);
      __hip_atomic_store(&bar[1], gen + 1u, __ATOMIC_RELEASE, __HIP_MEMORY_SCOPE_AGENT);
    } else {
      while (__hip_atomic_load(&bar[1], __ATOMIC_ACQUIRE, __HIP_MEMORY_SCOPE_AGENT) == gen)
        __builtin_amdgcn_s_sleep(1);
    }
  }
  __syncthreads();
  __threadfence();
}

// =====================================================================
// Kernel 1: prep — f16 weight packing, bias sums, state init, barrier init
// =====================================================================
__global__ void dec_prep_kernel(
    const int* __restrict__ trg, const float* __restrict__ h0in, const float* __restrict__ c0in,
    const float* __restrict__ enc, const float* __restrict__ embed,
    const float* __restrict__ Wenc,
    const float* __restrict__ Wih0, const float* __restrict__ Whh0,
    const float* __restrict__ bih0, const float* __restrict__ bhh0,
    const float* __restrict__ Wih1, const float* __restrict__ Whh1,
    const float* __restrict__ bih1, const float* __restrict__ bhh1,
    const float* __restrict__ Wlog,
    _Float16* __restrict__ enc16, _Float16* __restrict__ wenc16,
    _Float16* __restrict__ w0c, _Float16* __restrict__ w1c, _Float16* __restrict__ wlog16,
    float* __restrict__ b0c, float* __restrict__ b1c,
    float* __restrict__ h0s, float* __restrict__ c0s, float* __restrict__ h1s, float* __restrict__ c1s,
    _Float16* __restrict__ x0, unsigned* __restrict__ bar)
{
  const size_t tid = (size_t)blockIdx.x * blockDim.x + threadIdx.x;
  const size_t nth = (size_t)gridDim.x * blockDim.x;

  for (size_t i = tid; i < (size_t)V_ * (2 * H_); i += nth) wlog16[i] = (_Float16)Wlog[i];
  for (size_t i = tid; i < (size_t)B_ * S_ * H_;   i += nth) enc16[i]  = (_Float16)enc[i];
  for (size_t i = tid; i < (size_t)H_ * H_;        i += nth) wenc16[i] = (_Float16)Wenc[i];

  for (size_t i = tid; i < (size_t)G_ * K0C_; i += nth) {
    int n = (int)(i / K0C_), k = (int)(i % K0C_);
    float v;
    if (k < E_)          v = Wih0[(size_t)n * E_ + k];
    else if (k < EPAD_)  v = 0.0f;
    else                 v = Whh0[(size_t)n * H_ + (k - EPAD_)];
    w0c[i] = (_Float16)v;
  }
  for (size_t i = tid; i < (size_t)G_ * K1C_; i += nth) {
    int n = (int)(i / K1C_), k = (int)(i % K1C_);
    float v = (k < H_) ? Wih1[(size_t)n * H_ + k] : Whh1[(size_t)n * H_ + (k - H_)];
    w1c[i] = (_Float16)v;
  }
  for (size_t i = tid; i < (size_t)G_; i += nth) { b0c[i] = bih0[i] + bhh0[i]; b1c[i] = bih1[i] + bhh1[i]; }
  for (size_t i = tid; i < (size_t)B_ * H_; i += nth) {
    h0s[i] = h0in[i];             c0s[i] = c0in[i];
    h1s[i] = h0in[B_ * H_ + i];   c1s[i] = c0in[B_ * H_ + i];
  }
  // Initial X0 = [embed(t=0) | pad | h0(layer0)]
  for (size_t i = tid; i < (size_t)B_ * K0C_; i += nth) {
    int b = (int)(i / K0C_), k = (int)(i % K0C_);
    float v;
    if (k < E_)          v = embed[(size_t)trg[b * T_ + 0] * E_ + k];
    else if (k < EPAD_)  v = 0.0f;
    else                 v = h0in[(size_t)b * H_ + (k - EPAD_)];
    x0[i] = (_Float16)v;
  }
  if (tid == 0) { bar[0] = 0u; bar[1] = 0u; }
}

// =====================================================================
// Kernel 2: generic WMMA GEMM, 1x4 N-blocking per wave.
// C[M,N] = A[M,K](f16,row-major) * W[N,K]^T(f16,row-major) + bias[N]
// =====================================================================
__global__ void __launch_bounds__(256) dec_gemm_n4_kernel(
    const _Float16* __restrict__ A, const _Float16* __restrict__ W,
    const float* __restrict__ bias, float* __restrict__ C,
    int M, int N, int K)
{
  const int wpb  = blockDim.x >> 5;
  const int wave = blockIdx.x * wpb + (threadIdx.x >> 5);
  const int nwav = gridDim.x * wpb;
  const int mtiles = M >> 4, ngroups = N >> 6;
  const int jobs = mtiles * ngroups;
  const int lane = threadIdx.x & 31;

  for (int job = wave; job < jobs; job += nwav) {
    const int mt = job % mtiles;           // m fastest: concurrent waves share n-group -> W reuse in L2
    const int ng = job / mtiles;
    const int m0 = mt << 4, n0 = ng << 6;
    v8f a0 = {}, a1 = {}, a2 = {}, a3 = {};
    for (int k = 0; k < K; k += 32) {
      v16h a = load_a16(A, K, m0, k);
      a0 = wmma16(a, load_b16(W, K, n0,      k), a0);
      a1 = wmma16(a, load_b16(W, K, n0 + 16, k), a1);
      a2 = wmma16(a, load_b16(W, K, n0 + 32, k), a2);
      a3 = wmma16(a, load_b16(W, K, n0 + 48, k), a3);
    }
    const int n  = n0 + (lane & 15);
    const int mb = m0 + ((lane >> 4) << 3);
    const float bb0 = bias ? bias[n]      : 0.0f;
    const float bb1 = bias ? bias[n + 16] : 0.0f;
    const float bb2 = bias ? bias[n + 32] : 0.0f;
    const float bb3 = bias ? bias[n + 48] : 0.0f;
    CV c0, c1, c2, c3; c0.v = a0; c1.v = a1; c2.v = a2; c3.v = a3;
#pragma unroll
    for (int r = 0; r < 8; ++r) {
      float* row = C + (size_t)(mb + r) * N;
      row[n]      = c0.f[r] + bb0;
      row[n + 16] = c1.f[r] + bb1;
      row[n + 32] = c2.f[r] + bb2;
      row[n + 48] = c3.f[r] + bb3;
    }
  }
}

// =====================================================================
// Kernel 3: persistent recurrence (LSTM0 -> LSTM1 -> attention) over T
// =====================================================================
__global__ void __launch_bounds__(256) dec_recur_kernel(
    const int* __restrict__ trg, const float* __restrict__ embed,
    const unsigned char* __restrict__ mask, const float* __restrict__ memories,
    const _Float16* __restrict__ w0c, const _Float16* __restrict__ w1c,
    const float* __restrict__ b0c, const float* __restrict__ b1c,
    float* __restrict__ h0s, float* __restrict__ c0s,
    float* __restrict__ h1s, float* __restrict__ c1s,
    _Float16* __restrict__ x0, _Float16* __restrict__ x1,
    float* __restrict__ g0, float* __restrict__ g1,
    _Float16* __restrict__ abuf, unsigned* __restrict__ bar)
{
  const int tid  = threadIdx.x;
  const int wave = (blockIdx.x << 3) + (tid >> 5);  // 8 waves/block
  const int nwav = NBLK_ * 8;
  const int gi   = blockIdx.x * 256 + tid;          // 0..16383 == B_*H_

  __shared__ float sh_h1[H_];
  __shared__ float sh_sc[S_];
  __shared__ float sred[256];

  for (int t = 0; t < T_; ++t) {
    // gates0 = X0 @ W0cat^T   [32 x 2048], K=832
    for (int job = wave; job < 2 * 128; job += nwav)
      gemm_tile(x0, K0C_, w0c, K0C_, g0, G_, (job >> 7) << 4, (job & 127) << 4, K0C_);
    gbarrier(bar, NBLK_);

    // LSTM cell0 elementwise + build X1 = [h0n | h1_prev]
    {
      const int b = gi >> 9, h = gi & 511;
      const float* gr = g0 + (size_t)b * G_;
      float i_ = gr[h]          + b0c[h];
      float f_ = gr[512 + h]    + b0c[512 + h];
      float gg = gr[1024 + h]   + b0c[1024 + h];
      float o_ = gr[1536 + h]   + b0c[1536 + h];
      float c  = sigmf(f_) * c0s[gi] + sigmf(i_) * tanhf(gg);
      float hn = sigmf(o_) * tanhf(c);
      c0s[gi] = c; h0s[gi] = hn;
      x1[(size_t)b * K1C_ + h]       = (_Float16)hn;
      x1[(size_t)b * K1C_ + 512 + h] = (_Float16)h1s[gi];
    }
    gbarrier(bar, NBLK_);

    // gates1 = X1 @ W1cat^T   [32 x 2048], K=1024
    for (int job = wave; job < 2 * 128; job += nwav)
      gemm_tile(x1, K1C_, w1c, K1C_, g1, G_, (job >> 7) << 4, (job & 127) << 4, K1C_);
    gbarrier(bar, NBLK_);

    // LSTM cell1 elementwise -> h1n; store h1n half of A row
    {
      const int b = gi >> 9, h = gi & 511;
      const float* gr = g1 + (size_t)b * G_;
      float i_ = gr[h]          + b1c[h];
      float f_ = gr[512 + h]    + b1c[512 + h];
      float gg = gr[1024 + h]   + b1c[1024 + h];
      float o_ = gr[1536 + h]   + b1c[1536 + h];
      float c  = sigmf(f_) * c1s[gi] + sigmf(i_) * tanhf(gg);
      float hn = sigmf(o_) * tanhf(c);
      c1s[gi] = c; h1s[gi] = hn;
      abuf[(size_t)(b * T_ + t) * (2 * H_) + h] = (_Float16)hn;
    }
    // blocks 32..35 build X0 for t+1 (needs only h0s, stable since cell0)
    if (t + 1 < T_ && blockIdx.x >= 32 && blockIdx.x < 36) {
      for (int i = (blockIdx.x - 32) * 256 + tid; i < B_ * K0C_; i += 1024) {
        int b = i / K0C_, k = i % K0C_;
        float v;
        if (k < E_)          v = embed[(size_t)trg[b * T_ + t + 1] * E_ + k];
        else if (k < EPAD_)  v = 0.0f;
        else                 v = h0s[(size_t)b * H_ + (k - EPAD_)];
        x0[i] = (_Float16)v;
      }
    }
    gbarrier(bar, NBLK_);

    // Attention for step t: block b handles batch b (fp32 VALU; overlaps next step's GEMM)
    if (blockIdx.x < B_) {
      const int b = blockIdx.x;
      for (int h = tid; h < H_; h += 256) sh_h1[h] = h1s[(size_t)b * H_ + h];
      __syncthreads();
      for (int s = tid; s < S_; s += 256) {
        float e;
        if (mask[(size_t)b * S_ + s]) {
          e = NEGV;
        } else {
          const float* mrow = memories + ((size_t)b * S_ + s) * H_;
          float acc = 0.0f;
#pragma unroll 4
          for (int h = 0; h < H_; ++h) acc += sh_h1[h] * mrow[h];
          e = acc;
        }
        sh_sc[s] = e;
      }
      __syncthreads();
      float e0 = sh_sc[tid], e1 = sh_sc[tid + 256];
      sred[tid] = fmaxf(e0, e1);
      __syncthreads();
      for (int off = 128; off > 0; off >>= 1) { if (tid < off) sred[tid] = fmaxf(sred[tid], sred[tid + off]); __syncthreads(); }
      const float mx = sred[0];
      __syncthreads();
      float p0 = __expf(e0 - mx), p1 = __expf(e1 - mx);
      sh_sc[tid] = p0; sh_sc[tid + 256] = p1;
      sred[tid] = p0 + p1;
      __syncthreads();
      for (int off = 128; off > 0; off >>= 1) { if (tid < off) sred[tid] += sred[tid + off]; __syncthreads(); }
      const float inv = 1.0f / sred[0];
      for (int h = tid; h < H_; h += 256) {
        float acc = 0.0f;
        const float* mcol = memories + (size_t)b * S_ * H_ + h;
#pragma unroll 4
        for (int s = 0; s < S_; ++s) acc += sh_sc[s] * mcol[(size_t)s * H_];
        abuf[(size_t)(b * T_ + t) * (2 * H_) + H_ + h] = (_Float16)(acc * inv);
      }
      __syncthreads();
    }
  }
}

// =====================================================================
extern "C" void kernel_launch(void* const* d_in, const int* in_sizes, int n_in,
                              void* d_out, int out_size, void* d_ws, size_t ws_size,
                              hipStream_t stream) {
  const int*   trg   = (const int*)  d_in[0];
  const float* h0in  = (const float*)d_in[1];
  const float* c0in  = (const float*)d_in[2];
  const float* enc   = (const float*)d_in[3];
  const unsigned char* msk = (const unsigned char*)d_in[4];   // jax bool = 1 byte
  const float* embed = (const float*)d_in[5];
  const float* Wenc  = (const float*)d_in[6];
  const float* benc  = (const float*)d_in[7];
  const float* Wih0  = (const float*)d_in[8];
  const float* Whh0  = (const float*)d_in[9];
  const float* bih0  = (const float*)d_in[10];
  const float* bhh0  = (const float*)d_in[11];
  const float* Wih1  = (const float*)d_in[12];
  const float* Whh1  = (const float*)d_in[13];
  const float* bih1  = (const float*)d_in[14];
  const float* bhh1  = (const float*)d_in[15];
  const float* Wlog  = (const float*)d_in[16];
  const float* blog  = (const float*)d_in[17];
  float* out = (float*)d_out;

  char* ws = (char*)d_ws;
  size_t off = 0;
  auto alloc = [&](size_t bytes) { char* p = ws + off; off += (bytes + 255) & ~(size_t)255; return p; };

  float*    memories = (float*)   alloc((size_t)B_ * S_ * H_ * 4);
  _Float16* enc16    = (_Float16*)alloc((size_t)B_ * S_ * H_ * 2);
  _Float16* wlog16   = (_Float16*)alloc((size_t)V_ * 2 * H_ * 2);
  _Float16* wenc16   = (_Float16*)alloc((size_t)H_ * H_ * 2);
  _Float16* w0c      = (_Float16*)alloc((size_t)G_ * K0C_ * 2);
  _Float16* w1c      = (_Float16*)alloc((size_t)G_ * K1C_ * 2);
  _Float16* abuf     = (_Float16*)alloc((size_t)T_ * B_ * 2 * H_ * 2);
  _Float16* x0       = (_Float16*)alloc((size_t)B_ * K0C_ * 2);
  _Float16* x1       = (_Float16*)alloc((size_t)B_ * K1C_ * 2);
  float*    g0       = (float*)   alloc((size_t)B_ * G_ * 4);
  float*    g1       = (float*)   alloc((size_t)B_ * G_ * 4);
  float*    h0s      = (float*)   alloc((size_t)B_ * H_ * 4);
  float*    c0s      = (float*)   alloc((size_t)B_ * H_ * 4);
  float*    h1s      = (float*)   alloc((size_t)B_ * H_ * 4);
  float*    c1s      = (float*)   alloc((size_t)B_ * H_ * 4);
  float*    b0c      = (float*)   alloc((size_t)G_ * 4);
  float*    b1c      = (float*)   alloc((size_t)G_ * 4);
  unsigned* bar      = (unsigned*)alloc(256);
  (void)ws_size; (void)in_sizes; (void)n_in; (void)out_size;

  // 1) prep: f16 packing + state/barrier init
  dec_prep_kernel<<<1024, 256, 0, stream>>>(
      trg, h0in, c0in, enc, embed, Wenc, Wih0, Whh0, bih0, bhh0,
      Wih1, Whh1, bih1, bhh1, Wlog,
      enc16, wenc16, w0c, w1c, wlog16, b0c, b1c,
      h0s, c0s, h1s, c1s, x0, bar);

  // 2) memories = enc @ Wenc^T + benc  -> [16384, 512] f32
  dec_gemm_n4_kernel<<<512, 256, 0, stream>>>(enc16, wenc16, benc, memories,
                                              B_ * S_, H_, H_);

  // 3) persistent recurrence over T (WMMA gate GEMMs + attention)
  dec_recur_kernel<<<NBLK_, 256, 0, stream>>>(
      trg, embed, msk, memories, w0c, w1c, b0c, b1c,
      h0s, c0s, h1s, c1s, x0, x1, g0, g1, abuf, bar);

  // 4) logits = [h1|ctx] @ Wlog^T + blog -> [2048, 32000] f32 == [B,T,V]
  dec_gemm_n4_kernel<<<4000, 256, 0, stream>>>(abuf, wlog16, blog, out,
                                               T_ * B_, V_, 2 * H_);
}
